// Block_36447092474456
// MI455X (gfx1250) — matrix operand order repaired and checked by reading, not compiled, not run on hardware
//
#include <hip/hip_runtime.h>
#include <hip/hip_bf16.h>
#include <math.h>

// ---------------------------------------------------------------------------
// Transformer block on MI455X (gfx1250): all GEMMs via V_WMMA_F32_16X16X32_F16
// (f16 inputs, f32 accumulate), flash-attention with online softmax.
// GEMM waves are register-blocked 2x2 (32x32 output per wave).
// Attention K-tiles are streamed global->LDS with the Tensor Data Mover
// (tensor_load_to_lds, TENSORcnt) and double-buffered so the DMA of chunk
// j+1 overlaps softmax+WMMA of chunk j.
// ---------------------------------------------------------------------------

typedef __attribute__((ext_vector_type(16))) _Float16 v16h;
typedef __attribute__((ext_vector_type(8)))  _Float16 v8h;
typedef __attribute__((ext_vector_type(8)))  float    v8f;

union ABFrag { v16h v; v8h h[2]; };

__device__ __forceinline__ v8f wmma_f16(v16h a, v16h b, v8f c) {
  return __builtin_amdgcn_wmma_f32_16x16x32_f16(
      /*neg_a=*/false, a, /*neg_b=*/false, b,
      /*c_mod=*/(short)0, c, /*reuse_a=*/false, /*reuse_b=*/false);
}

__device__ __forceinline__ float rmax16(float v) {
  #pragma unroll
  for (int o = 8; o > 0; o >>= 1) v = fmaxf(v, __shfl_xor(v, o, 16));
  return v;
}
__device__ __forceinline__ float rsum16(float v) {
  #pragma unroll
  for (int o = 8; o > 0; o >>= 1) v += __shfl_xor(v, o, 16);
  return v;
}

// ---------------------------------------------------------------------------
// Tensor Data Mover: async 2D tile load global -> LDS (CDNA5 TDM, TENSORcnt).
// Descriptor packing per CDNA5 ISA ch.8 (D# groups 0..3).
// This toolchain exposes the 6-arg builtin:
//   (u32x4 g0, i32x8 g1, i32x4 g2, i32x4 g3, i32x8 ext, i32 cpol)
// ---------------------------------------------------------------------------
#if __has_builtin(__builtin_amdgcn_tensor_load_to_lds)
#define HAVE_TDM 1
typedef unsigned int tdm_u32x4 __attribute__((ext_vector_type(4)));
typedef int          tdm_i32x4 __attribute__((ext_vector_type(4)));
typedef int          tdm_i32x8 __attribute__((ext_vector_type(8)));

// Load a ROWS x COLS tile of f16 (cols contiguous, row stride = ROW_STRIDE
// elements) from global memory into LDS at ldst. Wave-uniform operands.
template <int COLS, int ROWS, int ROW_STRIDE>
__device__ __forceinline__ void tdm_load_tile_f16(const _Float16* gsrc,
                                                  _Float16* ldst) {
  unsigned long long ga = (unsigned long long)(size_t)gsrc;
  unsigned int galo = __builtin_amdgcn_readfirstlane((unsigned int)ga);
  unsigned int gahi = __builtin_amdgcn_readfirstlane((unsigned int)(ga >> 32));
  unsigned int la   = __builtin_amdgcn_readfirstlane((unsigned int)(size_t)ldst);
  // group0: count=1 | lds_addr[63:32] | global_addr[120:64] | type=2[127:126]
  tdm_u32x4 g0 = { 1u, la, galo, (gahi & 0x01ffffffu) | 0x80000000u };
  // group1: data_size=2B; tensor_dim0=COLS; tensor_dim1=ROWS;
  //         tile_dim0=COLS; tile_dim1=ROWS; dim0_stride=ROW_STRIDE (elems)
  tdm_i32x8 g1 = { (int)(1u << 16),
                   (int)((unsigned)COLS << 16),
                   (int)((unsigned)ROWS << 16),
                   (int)((unsigned)COLS << 16),
                   (int)(unsigned)ROWS,
                   (int)(unsigned)ROW_STRIDE,
                   0, 0 };
  tdm_i32x4 gz4 = { 0, 0, 0, 0 };
  tdm_i32x8 gz8 = { 0, 0, 0, 0, 0, 0, 0, 0 };
  __builtin_amdgcn_tensor_load_to_lds(g0, g1, gz4, gz4, gz8, 0);
}
#else
#define HAVE_TDM 0
#endif

// ---------------------------------------------------------------------------
// Weight convert + transpose: Wt[n*K + k] = (f16) W[k*N + n]
// ---------------------------------------------------------------------------
__global__ void wcvt_kernel(const float* __restrict__ W, _Float16* __restrict__ Wt,
                            int K, int N) {
  size_t idx = (size_t)blockIdx.x * 256 + threadIdx.x;
  size_t total = (size_t)K * N;
  if (idx >= total) return;
  int k = (int)(idx / N);
  int n = (int)(idx % N);
  Wt[(size_t)n * K + k] = (_Float16)W[idx];
}

// ---------------------------------------------------------------------------
// LayerNorm over last dim (C), output f16. One block (256 thr) per row.
// ---------------------------------------------------------------------------
__global__ __launch_bounds__(256)
void ln_kernel(const float* __restrict__ x, const float* __restrict__ gamma,
               const float* __restrict__ beta, _Float16* __restrict__ out, int C) {
  __shared__ float red[512];
  const int row = blockIdx.x;
  const int tid = threadIdx.x;
  const float* xr = x + (size_t)row * C;
  float s = 0.f, s2 = 0.f;
  for (int c = tid; c < C; c += 256) { float v = xr[c]; s += v; s2 += v * v; }
  red[tid] = s; red[256 + tid] = s2;
  __syncthreads();
  for (int st = 128; st > 0; st >>= 1) {
    if (tid < st) { red[tid] += red[tid + st]; red[256 + tid] += red[256 + tid + st]; }
    __syncthreads();
  }
  const float mu  = red[0] / (float)C;
  const float var = red[256] / (float)C - mu * mu;
  const float inv = rsqrtf(var + 1e-6f);
  for (int c = tid; c < C; c += 256) {
    float v = (xr[c] - mu) * inv * gamma[c] + beta[c];
    out[(size_t)row * C + c] = (_Float16)v;
  }
}

// ---------------------------------------------------------------------------
// WMMA GEMM: C[M,N] = act(alpha * A[M,K] @ B[K,N] + bias + resid)
//   A: f16 row-major [M,K];  Bt: f16 row-major [N,K] (pre-transposed B)
//   8 waves per block (2x4); each wave computes a 32x32 output (2x2 WMMA
//   tiles) -> block tile 64x128. 4 WMMAs per K-step per wave.
// ---------------------------------------------------------------------------
__global__ __launch_bounds__(256)
void gemm_kernel(const _Float16* __restrict__ A, const _Float16* __restrict__ Bt,
                 const float* __restrict__ bias, const float* __restrict__ resid,
                 float* __restrict__ outf, _Float16* __restrict__ outh,
                 int M, int N, int K, float alpha, int gelu) {
  const int w    = threadIdx.x >> 5;
  const int lane = threadIdx.x & 31;
  const int row0 = blockIdx.y * 64  + (w >> 2) * 32;
  const int col0 = blockIdx.x * 128 + (w & 3) * 32;
  const int g = lane >> 4;      // half-wave group
  const int r = lane & 15;      // row (A) / column (B) owned by this lane

  const _Float16* ap0 = A  + (size_t)(row0 + r)      * K + g * 16;
  const _Float16* ap1 = A  + (size_t)(row0 + 16 + r) * K + g * 16;
  const _Float16* bp0 = Bt + (size_t)(col0 + r)      * K + g * 16;
  const _Float16* bp1 = Bt + (size_t)(col0 + 16 + r) * K + g * 16;

  v8f acc00 = {}, acc01 = {}, acc10 = {}, acc11 = {};
  for (int k0 = 0; k0 < K; k0 += 32) {
    ABFrag a0, a1;
    a0.h[0] = *(const v8h*)(ap0 + k0);        // K = k0 + g*16 .. +7
    a0.h[1] = *(const v8h*)(ap0 + k0 + 8);    // K = k0 + g*16 + 8 .. +15
    a1.h[0] = *(const v8h*)(ap1 + k0);
    a1.h[1] = *(const v8h*)(ap1 + k0 + 8);
    v16h b0 = *(const v16h*)(bp0 + k0);       // K = k0 + g*16 .. +15
    v16h b1 = *(const v16h*)(bp1 + k0);
    acc00 = wmma_f16(a0.v, b0, acc00);
    acc01 = wmma_f16(a0.v, b1, acc01);
    acc10 = wmma_f16(a1.v, b0, acc10);
    acc11 = wmma_f16(a1.v, b1, acc11);
  }

  // Epilogue. C/D layout: VGPR i -> row = tile_row + i + 8*g, col = tile_col + r
  v8f accs[4] = {acc00, acc01, acc10, acc11};
  #pragma unroll
  for (int t = 0; t < 4; ++t) {
    const int ra = t >> 1, cb = t & 1;
    #pragma unroll
    for (int i = 0; i < 8; ++i) {
      const int rr = row0 + ra * 16 + i + 8 * g;
      const int cc = col0 + cb * 16 + r;
      float v = alpha * accs[t][i];
      if (bias)  v += bias[cc];
      if (resid) v += resid[(size_t)rr * N + cc];
      if (gelu)  v = 0.5f * v * (1.0f + erff(v * 0.70710678118654752f));
      if (outf)  outf[(size_t)rr * N + cc] = v;
      if (outh)  outh[(size_t)rr * N + cc] = (_Float16)v;
    }
  }
}

// ---------------------------------------------------------------------------
// Flash attention: q,k,v f16 stored as [b, n, h, d] = row-major [B*N, C].
// One wave handles one 16-query tile for one (b,h); streams keys in chunks
// of 32. LDS buffers are wave-private -> per-wave DS in-order, no barriers.
// K chunks are double-buffered via TDM async loads (TENSORcnt).
// ---------------------------------------------------------------------------
#define ATT_N 2048
#define ATT_C 1024
#define ATT_H 16
#define ATT_D 64

__global__ __launch_bounds__(128)
void attn_kernel(const _Float16* __restrict__ q, const _Float16* __restrict__ k,
                 const _Float16* __restrict__ v, _Float16* __restrict__ y) {
  // per-wave LDS: 2x K chunk (32x64), V^T chunk (64x32), P tile (16x32)
  __shared__ _Float16 lds[4 * 6656];
  const int w    = threadIdx.x >> 5;
  const int lane = threadIdx.x & 31;
  _Float16* kbuf0 = lds + w * 6656;       // [key][d]   32*64 (buffer 0)
  _Float16* kbuf1 = kbuf0 + 2048;         // [key][d]   32*64 (buffer 1)
  _Float16* vtbuf = kbuf1 + 2048;         // [d][key]   64*32
  _Float16* pbuf  = vtbuf + 2048;         // [qrow][key] 16*32

  const int gw = blockIdx.x * 4 + w;      // 0 .. 4095
  const int qt = gw & 127;                // query tile (N/16 = 128)
  const int bh = gw >> 7;
  const int hh = bh & (ATT_H - 1);
  const int bb = bh >> 4;

  const int g = lane >> 4;
  const int r = lane & 15;
  const size_t baseRow = (size_t)bb * ATT_N;
  const _Float16* kbase = k + baseRow * ATT_C + hh * ATT_D;

  // Q fragments (A operand), d in [0,32) and [32,64)
  const _Float16* qp = q + (baseRow + qt * 16 + r) * ATT_C + hh * ATT_D;
  ABFrag qa0, qa1;
  qa0.h[0] = *(const v8h*)(qp + g * 16);
  qa0.h[1] = *(const v8h*)(qp + 8 + g * 16);
  qa1.h[0] = *(const v8h*)(qp + 32 + g * 16);
  qa1.h[1] = *(const v8h*)(qp + 40 + g * 16);

  float m[8], l[8];
  v8f acc[4] = {{}, {}, {}, {}};
  #pragma unroll
  for (int i = 0; i < 8; ++i) { m[i] = -INFINITY; l[i] = 0.f; }

#if HAVE_TDM
  // prologue: async DMA of the first K chunk
  tdm_load_tile_f16<ATT_D, 32, ATT_C>(kbase, kbuf0);
#endif

  for (int j0 = 0; j0 < ATT_N; j0 += 32) {
    _Float16* kbuf = ((j0 >> 5) & 1) ? kbuf1 : kbuf0;
#if HAVE_TDM
    // issue DMA for the next chunk into the other buffer, then wait for the
    // current chunk (<=1 outstanding leaves only the prefetch in flight)
    if (j0 + 32 < ATT_N) {
      _Float16* knext = ((j0 >> 5) & 1) ? kbuf0 : kbuf1;
      tdm_load_tile_f16<ATT_D, 32, ATT_C>(kbase + (size_t)(j0 + 32) * ATT_C, knext);
      __builtin_amdgcn_s_wait_tensorcnt(1);
    } else {
      __builtin_amdgcn_s_wait_tensorcnt(0);
    }
#else
    // fallback: stage K chunk through VGPRs (each lane copies one key row)
    {
      const uint4* ks = (const uint4*)(kbase + (size_t)(j0 + lane) * ATT_C);
      uint4* kd = (uint4*)(kbuf + lane * 64);
      #pragma unroll
      for (int i = 0; i < 8; ++i) kd[i] = ks[i];
    }
#endif
    // ---- stage V chunk transposed: [d][key] (manual; TDM cannot transpose)
    {
      const uint4* vs = (const uint4*)(v + (baseRow + j0 + lane) * ATT_C + hh * ATT_D);
      uint4 tv[8];
      #pragma unroll
      for (int i = 0; i < 8; ++i) tv[i] = vs[i];
      const _Float16* ta = (const _Float16*)tv;
      #pragma unroll
      for (int i = 0; i < 64; ++i) vtbuf[i * 32 + lane] = ta[i];
    }

    // ---- scores S = Q K^T  (two 16x16 tiles, K-dim = d = 64)
    v8f s0 = {}, s1 = {};
    {
      v16h b0 = *(const v16h*)(kbuf + r * 64 + g * 16);
      v16h b1 = *(const v16h*)(kbuf + r * 64 + 32 + g * 16);
      s0 = wmma_f16(qa0.v, b0, s0);
      s0 = wmma_f16(qa1.v, b1, s0);
      v16h b2 = *(const v16h*)(kbuf + (16 + r) * 64 + g * 16);
      v16h b3 = *(const v16h*)(kbuf + (16 + r) * 64 + 32 + g * 16);
      s1 = wmma_f16(qa0.v, b2, s1);
      s1 = wmma_f16(qa1.v, b3, s1);
    }

    // ---- online softmax update (row = i + 8*g, cols across 16 lanes)
    #pragma unroll
    for (int i = 0; i < 8; ++i) {
      float mx = rmax16(fmaxf(s0[i], s1[i]));
      float mnew = fmaxf(m[i], mx);
      float corr = __expf(m[i] - mnew);
      float p0 = __expf(s0[i] - mnew);
      float p1 = __expf(s1[i] - mnew);
      float rs = rsum16(p0 + p1);
      l[i] = l[i] * corr + rs;
      m[i] = mnew;
      #pragma unroll
      for (int t = 0; t < 4; ++t) acc[t][i] = acc[t][i] * corr;
      s0[i] = p0; s1[i] = p1;
    }

    // ---- P -> LDS (C layout -> memory), reload as A fragment
    #pragma unroll
    for (int i = 0; i < 8; ++i) {
      pbuf[(i + 8 * g) * 32 + r]      = (_Float16)s0[i];
      pbuf[(i + 8 * g) * 32 + 16 + r] = (_Float16)s1[i];
    }
    ABFrag pa;
    pa.h[0] = *(const v8h*)(pbuf + r * 32 + g * 16);
    pa.h[1] = *(const v8h*)(pbuf + r * 32 + 8 + g * 16);

    // ---- O += P V   (N-dim = d, four 16-wide tiles; K-dim = 32 keys)
    #pragma unroll
    for (int t = 0; t < 4; ++t) {
      v16h vb = *(const v16h*)(vtbuf + (t * 16 + r) * 32 + g * 16);
      acc[t] = wmma_f16(pa.v, vb, acc[t]);
    }
  }

  // ---- normalize + store
  #pragma unroll
  for (int i = 0; i < 8; ++i) {
    float inv = (l[i] > 0.f) ? (1.f / l[i]) : 0.f;
    size_t orow = (baseRow + qt * 16 + i + 8 * g) * ATT_C + hh * ATT_D;
    #pragma unroll
    for (int t = 0; t < 4; ++t)
      y[orow + t * 16 + r] = (_Float16)(acc[t][i] * inv);
  }
}

// ---------------------------------------------------------------------------
// Host-side orchestration
// ---------------------------------------------------------------------------
extern "C" void kernel_launch(void* const* d_in, const int* in_sizes, int n_in,
                              void* d_out, int out_size, void* d_ws, size_t ws_size,
                              hipStream_t stream) {
  constexpr int Bb = 2, Nn = 2048, Cc = 1024, Ff = 4096;
  constexpr int BN = Bb * Nn;                // 4096 rows
  const float scale = 0.125f;                // (C/H)^-0.5 = 64^-0.5

  const float* x      = (const float*)d_in[0];
  const float* Wq     = (const float*)d_in[1];
  const float* Wk     = (const float*)d_in[2];
  const float* Wv     = (const float*)d_in[3];
  const float* Wp     = (const float*)d_in[4];
  const float* bp     = (const float*)d_in[5];
  const float* W1     = (const float*)d_in[6];
  const float* b1     = (const float*)d_in[7];
  const float* W2     = (const float*)d_in[8];
  const float* b2     = (const float*)d_in[9];
  const float* gamma1 = (const float*)d_in[10];
  const float* beta1  = (const float*)d_in[11];
  const float* gamma2 = (const float*)d_in[12];
  const float* beta2  = (const float*)d_in[13];
  float* out = (float*)d_out;

  // workspace carve-up (all sizes 256B-multiples)
  char* ws = (char*)d_ws;
  size_t off = 0;
  auto alloc = [&](size_t bytes) { char* p = ws + off; off += (bytes + 255) & ~(size_t)255; return p; };
  _Float16* Wq16 = (_Float16*)alloc((size_t)Cc * Cc * 2);
  _Float16* Wk16 = (_Float16*)alloc((size_t)Cc * Cc * 2);
  _Float16* Wv16 = (_Float16*)alloc((size_t)Cc * Cc * 2);
  _Float16* Wp16 = (_Float16*)alloc((size_t)Cc * Cc * 2);
  _Float16* W116 = (_Float16*)alloc((size_t)Cc * Ff * 2);
  _Float16* W216 = (_Float16*)alloc((size_t)Ff * Cc * 2);
  _Float16* xn16 = (_Float16*)alloc((size_t)BN * Cc * 2);
  _Float16* q16  = (_Float16*)alloc((size_t)BN * Cc * 2);
  _Float16* k16  = (_Float16*)alloc((size_t)BN * Cc * 2);
  _Float16* v16  = (_Float16*)alloc((size_t)BN * Cc * 2);
  _Float16* y16  = (_Float16*)alloc((size_t)BN * Cc * 2);
  float*    x1   = (float*)   alloc((size_t)BN * Cc * 4);
  _Float16* xn2  = (_Float16*)alloc((size_t)BN * Cc * 2);
  _Float16* h16  = (_Float16*)alloc((size_t)BN * Ff * 2);
  (void)ws_size; (void)n_in; (void)in_sizes; (void)out_size;

  // 1) weight convert + transpose -> f16 [N][K]
  {
    int tCC = Cc * Cc, tCF = Cc * Ff;
    wcvt_kernel<<<(tCC + 255) / 256, 256, 0, stream>>>(Wq, Wq16, Cc, Cc);
    wcvt_kernel<<<(tCC + 255) / 256, 256, 0, stream>>>(Wk, Wk16, Cc, Cc);
    wcvt_kernel<<<(tCC + 255) / 256, 256, 0, stream>>>(Wv, Wv16, Cc, Cc);
    wcvt_kernel<<<(tCC + 255) / 256, 256, 0, stream>>>(Wp, Wp16, Cc, Cc);
    wcvt_kernel<<<(tCF + 255) / 256, 256, 0, stream>>>(W1, W116, Cc, Ff);
    wcvt_kernel<<<(tCF + 255) / 256, 256, 0, stream>>>(W2, W216, Ff, Cc);
  }

  // 2) LN1
  ln_kernel<<<BN, 256, 0, stream>>>(x, gamma1, beta1, xn16, Cc);

  // 3) QKV projections (q pre-scaled)
  {
    dim3 grid(Cc / 128, BN / 64), block(256);
    gemm_kernel<<<grid, block, 0, stream>>>(xn16, Wq16, nullptr, nullptr, nullptr, q16,
                                            BN, Cc, Cc, scale, 0);
    gemm_kernel<<<grid, block, 0, stream>>>(xn16, Wk16, nullptr, nullptr, nullptr, k16,
                                            BN, Cc, Cc, 1.0f, 0);
    gemm_kernel<<<grid, block, 0, stream>>>(xn16, Wv16, nullptr, nullptr, nullptr, v16,
                                            BN, Cc, Cc, 1.0f, 0);
  }

  // 4) flash attention -> y16
  attn_kernel<<<1024, 128, 0, stream>>>(q16, k16, v16, y16);

  // 5) output projection + residual: x1 = x + y @ Wp + bp
  {
    dim3 grid(Cc / 128, BN / 64), block(256);
    gemm_kernel<<<grid, block, 0, stream>>>(y16, Wp16, bp, x, x1, nullptr,
                                            BN, Cc, Cc, 1.0f, 0);
  }

  // 6) LN2
  ln_kernel<<<BN, 256, 0, stream>>>(x1, gamma2, beta2, xn2, Cc);

  // 7) MLP up + exact GELU: h = gelu(xn2 @ W1 + b1)
  {
    dim3 grid(Ff / 128, BN / 64), block(256);
    gemm_kernel<<<grid, block, 0, stream>>>(xn2, W116, b1, nullptr, nullptr, h16,
                                            BN, Ff, Cc, 1.0f, 1);
  }

  // 8) MLP down + residual: out = x1 + h @ W2 + b2
  {
    dim3 grid(Cc / 128, BN / 64), block(256);
    gemm_kernel<<<grid, block, 0, stream>>>(h16, W216, b2, x1, out, nullptr,
                                            BN, Cc, Ff, 1.0f, 0);
  }
}
